// _PixelCNN_3178275799287
// MI455X (gfx1250) — matrix operand order repaired and checked by reading, not compiled, not run on hardware
//
#include <hip/hip_runtime.h>
#include <hip/hip_bf16.h>

typedef _Float16 v8h  __attribute__((ext_vector_type(8)));
typedef _Float16 v16h __attribute__((ext_vector_type(16)));
typedef float    v8f  __attribute__((ext_vector_type(8)));

#define HID   128
#define NL    5
#define IMG   64
#define BATCH 64

// ---------------------------------------------------------------------------
// Weight repack: wh [5][128cout][128cin][3][3] f32  ->  per-layer WMMA B-frags
// wrep layout: [L][tap(5)][ntile(8)][kc(4)][lane(32)][e(16)] f16
//   B-fragment (16x16x32 f16): lane l holds column N=l&15,
//   K = (l>=16 ? 16 : 0) + e   (identity K -> c_in mapping, chunk base kc*32)
// Tap t: (ky,kx) = t<3 ? (0,t) : (1,t-3)   == causal mask-B taps of the 3x3.
// ---------------------------------------------------------------------------
__global__ __launch_bounds__(256) void repack_wh(const float* __restrict__ wh,
                                                 _Float16* __restrict__ wrep) {
  int idx = blockIdx.x * 256 + threadIdx.x;          // total 5*81920
  int L  = idx / 81920; int r = idx % 81920;
  int t  = r / 16384;   r %= 16384;
  int nt = r / 2048;    r %= 2048;
  int kc = r / 512;     r %= 512;
  int lane = r / 16;    int e = r % 16;
  int ky = (t < 3) ? 0 : 1;
  int kx = (t < 3) ? t : (t - 3);
  int cout = nt * 16 + (lane & 15);
  int cin  = kc * 32 + ((lane & 16) ? 16 : 0) + e;
  float w = wh[(((size_t)(L * HID + cout) * HID + cin) * 3 + ky) * 3 + kx];
  wrep[idx] = (_Float16)w;
}

// ---------------------------------------------------------------------------
// Layer 0: 7x7 mask-A conv, 1 -> 128 channels, input x = 2*s-1, ReLU.
// Mask A active taps: rows ky=0..2 full, row ky=3 cols kx=0..2 (center excl.)
// Output NHWC f16: out[((b*64+y)*64+x)*128 + c]  (== linear thread index)
// ---------------------------------------------------------------------------
__global__ __launch_bounds__(256) void conv7x7_maskA(const float* __restrict__ s,
                                                     const float* __restrict__ w0,
                                                     const float* __restrict__ b0,
                                                     _Float16* __restrict__ out) {
  int idx = blockIdx.x * 256 + threadIdx.x;          // B*H*W*HID
  int c = idx & (HID - 1);
  int x = (idx >> 7)  & 63;
  int y = (idx >> 13) & 63;
  int b = idx >> 19;
  const float* sp = s + (size_t)b * IMG * IMG;
  const float* wp = w0 + c * 49;
  float acc = b0[c];
#pragma unroll
  for (int ky = 0; ky < 4; ++ky) {
    int yy = y + ky - 3;
    if (yy < 0) continue;
    const int kxmax = (ky < 3) ? 7 : 3;
#pragma unroll
    for (int kx = 0; kx < 7; ++kx) {
      if (kx >= kxmax) break;                        // folds at unroll time
      int xx = x + kx - 3;
      if (xx < 0 || xx >= IMG) continue;
      acc += wp[ky * 7 + kx] * (2.0f * sp[yy * IMG + xx] - 1.0f);
    }
  }
  out[idx] = (_Float16)fmaxf(acc, 0.0f);
}

// ---------------------------------------------------------------------------
// Hidden layer: 3x3 mask-B conv 128 -> 128 via implicit GEMM on WMMA.
// Block = 256 threads = 8 waves; one block owns 16 pixels (x0..x0+15 of row y)
// and each wave owns one 16-out-channel tile. 20 x v_wmma_f32_16x16x32_f16.
//
// Stage 1: cooperatively load the 2x18-pixel input halo (zero-filled at image
//          edges) into LDS once per block -> branch-free MMA main loop.
//          Channel dim padded 128->136 halves to spread LDS banks.
// Stage 2: per-wave WMMA loop; A from ds_load_b128 pairs, B from global b128.
// Stage 3: bias+ReLU into an LDS tile, then one coalesced global b128 store
//          per thread (instead of 8 scattered b16 stores per lane).
// ---------------------------------------------------------------------------
#define CPAD 136

__global__ __launch_bounds__(256) void conv3x3_maskB_wmma(
    const _Float16* __restrict__ in, const _Float16* __restrict__ wrep,
    const float* __restrict__ bias, _Float16* __restrict__ out) {
  __shared__ _Float16 sIn[2 * 18 * CPAD];            // 9,792 B
  __shared__ _Float16 sOut[16 * HID];                // 4,096 B

  const int tid  = threadIdx.x;
  const int lane = tid & 31;
  const int wv   = tid >> 5;                         // out-channel tile 0..7
  const int tile = blockIdx.x;
  const int x0 = (tile & 3) << 4;
  const int y  = (tile >> 2) & 63;
  const int b  = tile >> 8;

  // ---- Stage 1: halo load: row0 = y-1 (cols x0-1..x0+16), row1 = y ----
  for (int c = tid; c < 2 * 18 * 16; c += 256) {     // 16B chunks
    const int g    = c & 15;                         // channel group of 8
    const int colr = c >> 4;                         // 0..35
    const int col  = colr % 18;
    const int r    = colr / 18;
    const int xx   = x0 - 1 + col;
    const int yy   = (r == 0) ? (y - 1) : y;
    v8h v = {};
    if (yy >= 0 && xx >= 0 && xx < IMG) {
      v = *reinterpret_cast<const v8h*>(
          in + ((size_t)((b * IMG + yy) * IMG + xx) << 7) + g * 8);
    }
    *reinterpret_cast<v8h*>(&sIn[(r * 18 + col) * CPAD + g * 8]) = v;
  }
  __syncthreads();

  // ---- Stage 2: 20 WMMAs per wave ----
  const int mrow = lane & 15;                        // A-matrix row (pixel)
  const int kb8  = (lane & 16) ? 16 : 0;
  v8f acc = {};
#pragma unroll
  for (int t = 0; t < 5; ++t) {
    const int r  = (t < 3) ? 0 : 1;
    const int dx = (t < 3) ? (t - 1) : (t - 4);      // taps: (-1,{-1,0,1}),(0,{-1,0})
    const int abase = (r * 18 + (mrow + dx + 1)) * CPAD;
#pragma unroll
    for (int kc = 0; kc < 4; ++kc) {
      const int k0 = kc << 5;
      union { v16h v; v8h h[2]; } A, Bf;
      A.h[0] = *reinterpret_cast<const v8h*>(&sIn[abase + k0 + kb8]);
      A.h[1] = *reinterpret_cast<const v8h*>(&sIn[abase + k0 + 8 + kb8]);
      const _Float16* wp = wrep + ((((t * 8 + wv) * 4 + kc) * 32 + lane) << 4);
      Bf.h[0] = *reinterpret_cast<const v8h*>(wp);
      Bf.h[1] = *reinterpret_cast<const v8h*>(wp + 8);
      acc = __builtin_amdgcn_wmma_f32_16x16x32_f16(false, A.v, false, Bf.v,
                                                   (short)0, acc, false, false);
    }
  }

  // ---- Stage 3: bias + ReLU -> LDS transpose -> coalesced b128 stores ----
  // C/D layout: VGPR r holds (M = r + 8*(lane>=16), N = lane&15)
  const int n  = (wv << 4) + (lane & 15);
  const float bv = bias[n];
  const int mbase = (lane & 16) ? 8 : 0;
#pragma unroll
  for (int r2 = 0; r2 < 8; ++r2) {
    sOut[(mbase + r2) * HID + n] = (_Float16)fmaxf(acc[r2] + bv, 0.0f);
  }
  __syncthreads();
  {
    const int m = tid >> 4, g = tid & 15;
    _Float16* op = out + ((size_t)((b * IMG + y) * IMG + x0) << 7) + m * HID + g * 8;
    *reinterpret_cast<v8h*>(op) = *reinterpret_cast<const v8h*>(&sOut[m * HID + g * 8]);
  }
}

// ---------------------------------------------------------------------------
// Head: 1x1 conv 128->1 + Bernoulli log-prob, reduced over H,W per image.
// lp = s*logsig(z) + (1-s)*logsig(-z) = -( s*softplus(-z) + (1-s)*softplus(z) )
// with softplus(-z) = softplus(z) - z.
// ---------------------------------------------------------------------------
__global__ __launch_bounds__(256) void head_logprob(
    const _Float16* __restrict__ h, const float* __restrict__ s,
    const float* __restrict__ wout, const float* __restrict__ bout,
    float* __restrict__ outv) {
  __shared__ float red[256];
  const int b = blockIdx.x;
  const int tid = threadIdx.x;
  const float bz = bout[0];
  float lp = 0.0f;
  for (int p = tid; p < IMG * IMG; p += 256) {
    const _Float16* hp = h + ((size_t)(b * IMG * IMG + p) << 7);
    float z = bz;
#pragma unroll
    for (int c = 0; c < HID; c += 8) {
      v8h v = *reinterpret_cast<const v8h*>(hp + c);
#pragma unroll
      for (int j = 0; j < 8; ++j) z += (float)v[j] * wout[c + j];
    }
    float sv  = s[(size_t)b * IMG * IMG + p];
    float spp = fmaxf(z, 0.0f) + log1pf(__expf(-fabsf(z)));  // softplus(z)
    float spm = spp - z;                                     // softplus(-z)
    lp -= sv * spm + (1.0f - sv) * spp;
  }
  red[tid] = lp;
  __syncthreads();
  for (int off = 128; off > 0; off >>= 1) {
    if (tid < off) red[tid] += red[tid + off];
    __syncthreads();
  }
  if (tid == 0) outv[b] = red[0];
}

// ---------------------------------------------------------------------------
extern "C" void kernel_launch(void* const* d_in, const int* in_sizes, int n_in,
                              void* d_out, int out_size, void* d_ws, size_t ws_size,
                              hipStream_t stream) {
  const float* samples = (const float*)d_in[0];
  const float* w0    = (const float*)d_in[1];
  const float* b0    = (const float*)d_in[2];
  const float* wh    = (const float*)d_in[3];
  const float* bh    = (const float*)d_in[4];
  const float* w_out = (const float*)d_in[5];
  const float* b_out = (const float*)d_in[6];
  float* out = (float*)d_out;

  const size_t actElems = (size_t)BATCH * IMG * IMG * HID;   // 33,554,432 halves
  _Float16* hA   = (_Float16*)d_ws;
  _Float16* hB   = hA + actElems;
  _Float16* wrep = hB + actElems;                            // 5 * 81920 halves

  repack_wh<<<(5 * 81920) / 256, 256, 0, stream>>>(wh, wrep);
  conv7x7_maskA<<<(int)(actElems / 256), 256, 0, stream>>>(samples, w0, b0, hA);

  const _Float16* cin = hA;
  _Float16* cout = hB;
  for (int L = 0; L < NL; ++L) {
    conv3x3_maskB_wmma<<<BATCH * IMG * (IMG / 16), 256, 0, stream>>>(
        cin, wrep + (size_t)L * 81920, bh + L * HID, cout);
    _Float16* tmp = cout;
    cout = (_Float16*)cin;
    cin = tmp;
  }
  head_logprob<<<BATCH, 256, 0, stream>>>(cin, samples, w_out, b_out, out);
}
